// GraphGonvLayer_50775103373809
// MI455X (gfx1250) — matrix operand order repaired and checked by reading, not compiled, not run on hardware
//
#include <hip/hip_runtime.h>
#include <hip/hip_bf16.h>

// ---------------------------------------------------------------------------
// GraphConv layer for MI455X (gfx1250, wave32, WMMA).
// Sparse-softmax insight: adj has <=16 nonzeros/row, so only the similarity
// GEMM and the two output projections are dense matrix work. Both run on
// v_wmma_f32_16x16x32_f16 with hi/lo f16 error-splitting (~f32 fidelity,
// f32 accumulation). Everything is L2-resident (<< 192 MB), so operands are
// streamed straight from global memory.
// ---------------------------------------------------------------------------

typedef _Float16 v16h __attribute__((ext_vector_type(16)));
typedef float    v8f  __attribute__((ext_vector_type(8)));

#define BB 8
#define NN 2048
#define DW 512
#define KK 16
#define MROWS (BB * NN)   // 16384

// ---- WMMA helpers ---------------------------------------------------------

static __device__ __forceinline__ v8f wmma_f16(v16h a, v16h b, v8f c) {
  // 8 args: (neg_a, A, neg_b, B, c_mod, C, reuse_a, reuse_b)
  return __builtin_amdgcn_wmma_f32_16x16x32_f16(false, a, false, b, (short)0, c,
                                                false, false);
}

// A fragment (16x32 f16, M x K), per ISA table:
//   lanes 0-15: M=lane,   K groups {0..7, 16..23}
//   lanes16-31: M=lane-16,K groups {8..15, 24..31}
// `tile` points at row 0 of the 16-row tile, row stride DW.
static __device__ __forceinline__ v16h frag_a(const _Float16* __restrict__ tile,
                                              int lane, int k0) {
  const _Float16* p = tile + (size_t)(lane & 15) * DW + k0 + ((lane & 16) ? 8 : 0);
  v16h r;
#pragma unroll
  for (int e = 0; e < 8; ++e) { r[e] = p[e]; r[e + 8] = p[e + 16]; }
  return r;
}

// B fragment (32x16 f16, K x N): N = lane&15, K groups as A fragment.
// `bt` is a K-major (transposed) operand: bt[k*ld + n], contiguous in n.
static __device__ __forceinline__ v16h frag_b(const _Float16* __restrict__ bt,
                                              int ld, int lane, int k0, int n0) {
  const _Float16* p = bt + (size_t)(k0 + ((lane & 16) ? 8 : 0)) * ld + n0 + (lane & 15);
  v16h r;
#pragma unroll
  for (int e = 0; e < 8; ++e) { r[e] = p[(size_t)e * ld]; r[e + 8] = p[(size_t)(e + 16) * ld]; }
  return r;
}

// ---- Kernel 1: transpose + hi/lo split of a 512x512 weight ----------------
__global__ __launch_bounds__(256) void k_wsplit(const float* __restrict__ w,
                                                _Float16* __restrict__ th,
                                                _Float16* __restrict__ tl) {
  int gid = blockIdx.x * 256 + threadIdx.x;   // 512*512 threads
  int o = gid >> 9, d = gid & (DW - 1);
  float v = w[(size_t)o * DW + d];
  _Float16 h = (_Float16)v;
  th[(size_t)d * DW + o] = h;
  tl[(size_t)d * DW + o] = (_Float16)(v - (float)h);
}

// ---- Kernel 2: norms, x hi/lo splits (row-major + transposed), attn logits
__global__ __launch_bounds__(256) void k_prep(
    const float* __restrict__ x, const float* __restrict__ attn_w,
    const float* __restrict__ attn_b, _Float16* __restrict__ xh,
    _Float16* __restrict__ xl, _Float16* __restrict__ xth,
    _Float16* __restrict__ xtl, float* __restrict__ inv_norm,
    float* __restrict__ attn) {
  const int row = blockIdx.x;          // global row in [0, B*N)
  const int tid = threadIdx.x;         // 256
  const int b = row >> 11, n = row & (NN - 1);
  __shared__ float sx[DW];
  __shared__ float red[256];
  __shared__ float ar[KK][16];

  const float2 v = ((const float2*)(x + (size_t)row * DW))[tid];
  sx[2 * tid] = v.x; sx[2 * tid + 1] = v.y;
  red[tid] = v.x * v.x + v.y * v.y;
  __syncthreads();
  for (int s = 128; s > 0; s >>= 1) {
    if (tid < s) red[tid] += red[tid + s];
    __syncthreads();
  }
  if (tid == 0) inv_norm[row] = rsqrtf(red[0]);

  _Float16 h0 = (_Float16)v.x, h1 = (_Float16)v.y;
  _Float16 l0 = (_Float16)(v.x - (float)h0), l1 = (_Float16)(v.y - (float)h1);
  size_t o = (size_t)row * DW + 2 * tid;
  xh[o] = h0; xh[o + 1] = h1; xl[o] = l0; xl[o + 1] = l1;
  size_t t0 = ((size_t)b * DW + 2 * tid) * NN + n;
  size_t t1 = ((size_t)b * DW + 2 * tid + 1) * NN + n;
  xth[t0] = h0; xth[t1] = h1; xtl[t0] = l0; xtl[t1] = l1;

  // attn[row,g] = x_row . attn_w[g] + attn_b[g]  (16 logits, 16 threads each)
  const int g = tid >> 4, u = tid & 15;
  const float* wrow = attn_w + (size_t)g * DW + u * 32;
  const float* xs = sx + u * 32;
  float p = 0.f;
#pragma unroll
  for (int c = 0; c < 32; ++c) p = fmaf(xs[c], wrow[c], p);
  ar[g][u] = p;
  __syncthreads();
  if (u == 0) {
    float s = attn_b[g];
#pragma unroll
    for (int t = 0; t < 16; ++t) s += ar[g][t];
    attn[(size_t)row * KK + g] = s;
  }
}

// ---- Kernel 3: 16-row similarity strip (WMMA) + top-16 selection ----------
__global__ __launch_bounds__(256) void k_sim_topk(
    const _Float16* __restrict__ xh, const _Float16* __restrict__ xl,
    const _Float16* __restrict__ xth, const _Float16* __restrict__ xtl,
    const float* __restrict__ inv_norm, int* __restrict__ idx) {
  extern __shared__ float smem[];
  const int SJ = NN + 8;                      // pad to break bank alignment
  float* sims = smem;                         // [16][SJ]
  float* cv = smem + 16 * SJ;                 // [16][256] candidate values
  int* ci = (int*)(cv + 16 * 256);            // [16][256] candidate indices

  const int tid = threadIdx.x;
  const int wave = tid >> 5, lane = tid & 31;
  const int r0 = blockIdx.x * 16;             // row tile within batch
  const int b = blockIdx.y;

  const _Float16* ah_t = xh + ((size_t)b * NN + r0) * DW;
  const _Float16* al_t = xl + ((size_t)b * NN + r0) * DW;
  const _Float16* bh_t = xth + (size_t)b * DW * NN;
  const _Float16* bl_t = xtl + (size_t)b * DW * NN;
  const float* invb = inv_norm + (size_t)b * NN;

  // each wave owns 256 contiguous columns (16 j-tiles of 16)
  for (int jt = 0; jt < 16; ++jt) {
    const int j0 = wave * 256 + jt * 16;
    v8f c = {};
    for (int k0 = 0; k0 < DW; k0 += 32) {
      v16h ah = frag_a(ah_t, lane, k0);
      v16h al = frag_a(al_t, lane, k0);
      v16h bh = frag_b(bh_t, NN, lane, k0, j0);
      v16h bl = frag_b(bl_t, NN, lane, k0, j0);
      c = wmma_f16(ah, bh, c);                // hi*hi
      c = wmma_f16(ah, bl, c);                // hi*lo
      c = wmma_f16(al, bh, c);                // lo*hi
    }
    // scale by 1/||x_j|| (per-row 1/||x_i|| doesn't change the ranking)
    const float invj = invb[j0 + (lane & 15)];
    const int rr = (lane & 16) ? 8 : 0;       // C layout: M = e + rr, N = lane&15
    const int jj = j0 + (lane & 15);
#pragma unroll
    for (int e = 0; e < 8; ++e) sims[(size_t)(e + rr) * SJ + jj] = c[e] * invj;
  }
  __syncthreads();

  // Phase A: thread (r, cch) -> local top-16 over a 128-column chunk.
  const int r = tid >> 4, cch = tid & 15;
  float bv[16]; int bi[16];
#pragma unroll
  for (int s = 0; s < 16; ++s) { bv[s] = -3.4e38f; bi[s] = 0x7fffffff; }
  const float* srow = sims + (size_t)r * SJ;
  for (int jj = 0; jj < 128; ++jj) {
    int j = (cch << 7) | ((jj + cch * 4) & 127);   // bank-staggered scan
    float v = srow[j]; int id = j;
#pragma unroll
    for (int s = 0; s < 16; ++s) {                 // jax tie-break: idx asc
      bool better = (v > bv[s]) || ((v == bv[s]) && (id < bi[s]));
      float tv = better ? bv[s] : v; int ti = better ? bi[s] : id;
      bv[s] = better ? v : bv[s];    bi[s] = better ? id : bi[s];
      v = tv; id = ti;
    }
  }
#pragma unroll
  for (int s = 0; s < 16; ++s) {
    cv[r * 256 + cch * 16 + s] = bv[s];
    ci[r * 256 + cch * 16 + s] = bi[s];
  }
  __syncthreads();

  // Phase B: one thread per row merges its 256 candidates.
  if (tid < 16) {
    float fv[16]; int fi[16];
#pragma unroll
    for (int s = 0; s < 16; ++s) { fv[s] = -3.4e38f; fi[s] = 0x7fffffff; }
    const float* cvr = cv + tid * 256; const int* cir = ci + tid * 256;
    for (int q0 = 0; q0 < 256; ++q0) {
      int q = (q0 + tid * 4) & 255;                // bank-staggered
      float v = cvr[q]; int id = cir[q];
#pragma unroll
      for (int s = 0; s < 16; ++s) {
        bool better = (v > fv[s]) || ((v == fv[s]) && (id < fi[s]));
        float tv = better ? fv[s] : v; int ti = better ? fi[s] : id;
        fv[s] = better ? v : fv[s];    fi[s] = better ? id : fi[s];
        v = tv; id = ti;
      }
    }
    const size_t orow = (size_t)b * NN + r0 + tid;
#pragma unroll
    for (int s = 0; s < 16; ++s) idx[orow * KK + s] = fi[s];
  }
}

// ---- Kernel 4: mutual-knn mask + exact sparse softmax ---------------------
__global__ __launch_bounds__(256) void k_mask_softmax(
    const int* __restrict__ idx, const float* __restrict__ attn,
    float* __restrict__ adjw) {
  const int i = blockIdx.x * 256 + threadIdx.x;  // global row
  if (i >= MROWS) return;
  const int b = i >> 11, il = i & (NN - 1);
  const int* my = idx + (size_t)i * KK;
  const float* at = attn + (size_t)i * KK;
  float lg[16];
#pragma unroll
  for (int k = 0; k < KK; ++k) {
    const int j = my[k];
    const int* nb = idx + ((size_t)b * NN + j) * KK;
    bool mut = false;
#pragma unroll
    for (int t = 0; t < KK; ++t) mut |= (nb[t] == il);
    lg[k] = mut ? at[k] : -3.0e38f;
  }
  float m = lg[0];
#pragma unroll
  for (int k = 1; k < KK; ++k) m = fmaxf(m, lg[k]);
  float e[16], z = 0.f;
#pragma unroll
  for (int k = 0; k < KK; ++k) {
    e[k] = (lg[k] > -1.0e38f) ? expf(lg[k] - m) : 0.f;  // non-mutual -> exact 0
    z += e[k];
  }
  const float rz = 1.f / z;                      // diagonal always mutual: z>0
#pragma unroll
  for (int k = 0; k < KK; ++k) adjw[(size_t)i * KK + k] = e[k] * rz;
}

// ---- Kernel 5: sparse aggregation merged = sum_k adj*x_j, f16 split -------
__global__ __launch_bounds__(256) void k_aggregate(
    const float* __restrict__ x, const float* __restrict__ adjw,
    const int* __restrict__ idx, _Float16* __restrict__ mh,
    _Float16* __restrict__ ml) {
  const int row = blockIdx.x, tid = threadIdx.x;
  const int b = row >> 11;
  __shared__ float sw[KK];
  __shared__ int sj[KK];
  if (tid < KK) {
    sw[tid] = adjw[(size_t)row * KK + tid];
    sj[tid] = idx[(size_t)row * KK + tid];
  }
  __syncthreads();
  float ax = 0.f, ay = 0.f;
  const float2* xb = (const float2*)x;
#pragma unroll
  for (int k = 0; k < KK; ++k) {
    const float wgt = sw[k];                      // block-uniform branch
    if (wgt != 0.f) {
      const float2 xv = xb[((size_t)b * NN + sj[k]) * (DW / 2) + tid];
      ax = fmaf(wgt, xv.x, ax);
      ay = fmaf(wgt, xv.y, ay);
    }
  }
  const _Float16 h0 = (_Float16)ax, h1 = (_Float16)ay;
  const size_t o = (size_t)row * DW + 2 * tid;
  mh[o] = h0; mh[o + 1] = h1;
  ml[o] = (_Float16)(ax - (float)h0);
  ml[o + 1] = (_Float16)(ay - (float)h1);
}

// ---- Kernel 6: out = relu(merged@Wm^T) + x@Wo^T (WMMA, hi/lo split) -------
__global__ __launch_bounds__(256) void k_out_gemm(
    const _Float16* __restrict__ mh, const _Float16* __restrict__ ml,
    const _Float16* __restrict__ xh, const _Float16* __restrict__ xl,
    const _Float16* __restrict__ wmt_h, const _Float16* __restrict__ wmt_l,
    const _Float16* __restrict__ wot_h, const _Float16* __restrict__ wot_l,
    float* __restrict__ out) {
  const int tid = threadIdx.x, wave = tid >> 5, lane = tid & 31;
  const int m0 = blockIdx.x * 16;                // 16 rows per block
  const int n0 = blockIdx.y * 128 + wave * 16;   // 16 cols per wave
  const _Float16* am_h = mh + (size_t)m0 * DW;
  const _Float16* am_l = ml + (size_t)m0 * DW;
  const _Float16* ax_h = xh + (size_t)m0 * DW;
  const _Float16* ax_l = xl + (size_t)m0 * DW;
  v8f cm = {}, cx = {};
  for (int k0 = 0; k0 < DW; k0 += 32) {
    v16h amh = frag_a(am_h, lane, k0);
    v16h aml = frag_a(am_l, lane, k0);
    v16h axh = frag_a(ax_h, lane, k0);
    v16h axl = frag_a(ax_l, lane, k0);
    v16h bmh = frag_b(wmt_h, DW, lane, k0, n0);
    v16h bml = frag_b(wmt_l, DW, lane, k0, n0);
    v16h boh = frag_b(wot_h, DW, lane, k0, n0);
    v16h bol = frag_b(wot_l, DW, lane, k0, n0);
    cm = wmma_f16(amh, bmh, cm); cm = wmma_f16(amh, bml, cm); cm = wmma_f16(aml, bmh, cm);
    cx = wmma_f16(axh, boh, cx); cx = wmma_f16(axh, bol, cx); cx = wmma_f16(axl, boh, cx);
  }
  const int col = n0 + (lane & 15);
  const int rbase = m0 + ((lane & 16) ? 8 : 0);
#pragma unroll
  for (int e = 0; e < 8; ++e) {
    out[(size_t)(rbase + e) * DW + col] = fmaxf(cm[e], 0.f) + cx[e];
  }
}

// ---------------------------------------------------------------------------
extern "C" void kernel_launch(void* const* d_in, const int* in_sizes, int n_in,
                              void* d_out, int out_size, void* d_ws,
                              size_t ws_size, hipStream_t stream) {
  (void)in_sizes; (void)n_in; (void)out_size; (void)ws_size;
  const float* x        = (const float*)d_in[0];
  const float* attn_w   = (const float*)d_in[1];
  const float* attn_b   = (const float*)d_in[2];
  const float* w_merged = (const float*)d_in[3];
  const float* w_orig   = (const float*)d_in[4];
  float* out = (float*)d_out;

  // workspace partition (~105 MB total)
  char* p = (char*)d_ws;
  auto take = [&](size_t bytes) {
    char* q = p;
    p += (bytes + 255) & ~(size_t)255;
    return q;
  };
  const size_t XE = (size_t)BB * NN * DW;        // 8.39M elements
  _Float16* xh  = (_Float16*)take(XE * 2);
  _Float16* xl  = (_Float16*)take(XE * 2);
  _Float16* xth = (_Float16*)take(XE * 2);
  _Float16* xtl = (_Float16*)take(XE * 2);
  _Float16* mh  = (_Float16*)take(XE * 2);
  _Float16* ml  = (_Float16*)take(XE * 2);
  _Float16* wmt_h = (_Float16*)take((size_t)DW * DW * 2);
  _Float16* wmt_l = (_Float16*)take((size_t)DW * DW * 2);
  _Float16* wot_h = (_Float16*)take((size_t)DW * DW * 2);
  _Float16* wot_l = (_Float16*)take((size_t)DW * DW * 2);
  float* inv_norm = (float*)take((size_t)MROWS * 4);
  float* attn     = (float*)take((size_t)MROWS * KK * 4);
  int*   idx      = (int*)take((size_t)MROWS * KK * 4);
  float* adjw     = (float*)take((size_t)MROWS * KK * 4);

  k_wsplit<<<dim3(DW * DW / 256), 256, 0, stream>>>(w_merged, wmt_h, wmt_l);
  k_wsplit<<<dim3(DW * DW / 256), 256, 0, stream>>>(w_orig, wot_h, wot_l);
  k_prep<<<dim3(MROWS), 256, 0, stream>>>(x, attn_w, attn_b, xh, xl, xth, xtl,
                                          inv_norm, attn);
  const size_t lds = (size_t)(16 * (NN + 8) + 16 * 256) * sizeof(float) +
                     (size_t)(16 * 256) * sizeof(int);   // ~164 KB < 320 KB
  k_sim_topk<<<dim3(NN / 16, BB), 256, lds, stream>>>(xh, xl, xth, xtl,
                                                      inv_norm, idx);
  k_mask_softmax<<<dim3(MROWS / 256), 256, 0, stream>>>(idx, attn, adjw);
  k_aggregate<<<dim3(MROWS), 256, 0, stream>>>(x, adjw, idx, mh, ml);
  k_out_gemm<<<dim3(MROWS / 16, DW / 128), 256, 0, stream>>>(
      mh, ml, xh, xl, wmt_h, wmt_l, wot_h, wot_l, out);
}
